// GNN_2D_Encoder_43843026157657
// MI455X (gfx1250) — compile-verified, matrix operand order checked by reading, not run on hardware
//
#include <hip/hip_runtime.h>

#define N_NODES 10000
#define N_EDGES 160000
#define EF      170000      // E + N (self loops appended)
#define NODE_IN 64
#define EDGE_IN 16
#define HID     128
#define HEADS   4
#define HH      512         // HEADS*HID
#define NEGS    0.2f

typedef __attribute__((ext_vector_type(16))) __bf16 bf16x16;
typedef __attribute__((ext_vector_type(8)))  float  f32x8;
typedef __attribute__((ext_vector_type(4)))  float  f32x4;

// ---------------------------------------------------------------------------
// A-fragment loader: 16x32 bf16 tile of a row-major fp32 matrix.
// lane<16 : row = lane,    k = {kb..kb+7, kb+16..kb+23}
// lane>=16: row = lane-16, k = {kb+8..kb+15, kb+24..kb+31}
// Kdim must be a compile-time constant at the call site so the guard folds.
// ---------------------------------------------------------------------------
__device__ __forceinline__ bf16x16 load_a_frag(const float* __restrict__ rowPtr,
                                               int kb, int half, int Kdim) {
  bf16x16 a;
  int k0 = kb + 8 * half;
  int k1 = k0 + 16;
  if (k1 + 8 <= Kdim) {                       // folds when Kdim is constexpr
    f32x4 lo0 = *(const f32x4*)(rowPtr + k0);
    f32x4 lo1 = *(const f32x4*)(rowPtr + k0 + 4);
    f32x4 hi0 = *(const f32x4*)(rowPtr + k1);
    f32x4 hi1 = *(const f32x4*)(rowPtr + k1 + 4);
#pragma unroll
    for (int i = 0; i < 4; i++) {
      a[i]      = (__bf16)lo0[i];
      a[4 + i]  = (__bf16)lo1[i];
      a[8 + i]  = (__bf16)hi0[i];
      a[12 + i] = (__bf16)hi1[i];
    }
  } else {                                    // padded path (K=16 edge_W)
#pragma unroll
    for (int i = 0; i < 16; i++) {
      int k = k0 + ((i < 8) ? i : i + 8);
      a[i] = (k < Kdim) ? (__bf16)rowPtr[k] : (__bf16)0.0f;
    }
  }
  return a;
}

// ---------------------------------------------------------------------------
// Pack a fp32 weight matrix W[K x Ncols] into fragment-major bf16 so a lane's
// 16 B-values are one contiguous 32B vector: P[((kt*Ntiles+nt)*32+lane)*16+i]
// ---------------------------------------------------------------------------
__global__ void __launch_bounds__(32)
pack_b_bf16(const float* __restrict__ W, __bf16* __restrict__ P,
            int Kdim, int Ncols) {
  int kt = blockIdx.x, nt = blockIdx.y;
  int lane = threadIdx.x;
  int half = lane >> 4, r = lane & 15;
  int kb = kt * 32;
  int n = nt * 16 + r;
  bf16x16 frag;
#pragma unroll
  for (int i = 0; i < 16; i++) {
    int k = kb + 8 * half + ((i < 8) ? i : i + 8);
    float v = (k < Kdim) ? W[(size_t)k * Ncols + n] : 0.0f;
    frag[i] = (__bf16)v;
  }
  int Ntiles = Ncols >> 4;
  *(bf16x16*)(P + (((size_t)kt * Ntiles + nt) * 32 + lane) * 16) = frag;
}

// ---------------------------------------------------------------------------
// GEMM: C[M x NC] = A[M x KD](fp32) @ packedB(bf16) + bias.  KD/NC compile-
// time: branch-free, fully unrolled k-loop (KT*NT independent wmma+loads for
// the scheduler to pipeline).  One wave per 16(M) x 16*NT(N) tile; A frags
// reused NT times.  grid = (M/16, NC/(16*NT)), block = 32.
// ---------------------------------------------------------------------------
template <int KD, int NC, int NT>
__global__ void __launch_bounds__(32)
gemm_bias_wmma(const float* __restrict__ A,
               const __bf16* __restrict__ P,
               const float* __restrict__ bias,
               float* __restrict__ C, int M) {
  constexpr int KT = (KD + 31) / 32;
  constexpr int Ntiles = NC >> 4;
  int lane = threadIdx.x;
  int half = lane >> 4, r = lane & 15;
  int mb = blockIdx.x * 16;
  int nt0 = blockIdx.y * NT;
  f32x8 acc[NT] = {};
  const float* rowPtr = A + (size_t)(mb + r) * KD;
  const __bf16* pb = P + ((size_t)nt0 * 32 + lane) * 16;  // +kt*Ntiles*512 +j*512 are immediates
#pragma unroll
  for (int kt = 0; kt < KT; kt++) {
    bf16x16 a = load_a_frag(rowPtr, kt * 32, half, KD);
#pragma unroll
    for (int j = 0; j < NT; j++) {
      bf16x16 b = *(const bf16x16*)(pb + ((size_t)kt * Ntiles + j) * 512);
      acc[j] = __builtin_amdgcn_wmma_f32_16x16x32_bf16(false, a, false, b,
                                                       (short)0, acc[j], false, false);
    }
  }
#pragma unroll
  for (int j = 0; j < NT; j++) {
    int nb = (nt0 + j) * 16;
    float bv = bias[nb + r];
#pragma unroll
    for (int v = 0; v < 8; v++) {
      int m = mb + v + 8 * half;
      C[(size_t)m * NC + nb + r] = acc[j][v] + bv;
    }
  }
}

// ---------------------------------------------------------------------------
// Fused edge-logit kernel: for a 16-edge tile and one head, compute
//   ee = e_f @ We (WMMA, accumulators only — never materialized to memory),
//   m  = leaky_relu(xl[src] + xr[dst] + ee),
//   logit = sum_c m*att   (16-lane shuffle reduction)
// Gathers use 32-bit byte offsets against uniform XL/XR bases (saddr+voffset
// addressing; chunk offset is a compile-time immediate) to keep VGPR pressure
// below the spill threshold.  Two WMMA accumulators break the k-chain.
// grid = (EF/16, HEADS), block = 32
// ---------------------------------------------------------------------------
__global__ void __launch_bounds__(32)
edge_logits_wmma(const float* __restrict__ ef,
                 const __bf16* __restrict__ WeP,
                 const float* __restrict__ XL,
                 const float* __restrict__ XR,
                 const float* __restrict__ att,   // [HEADS*HID]
                 const int* __restrict__ src,
                 const int* __restrict__ dst,
                 float* __restrict__ logits) {
  int lane = threadIdx.x;
  int half = lane >> 4, r = lane & 15;
  int eb = blockIdx.x * 16;
  int head = blockIdx.y;

  bf16x16 afr[4];
  const float* rowPtr = ef + (size_t)(eb + r) * HID;
#pragma unroll
  for (int kt = 0; kt < 4; kt++) afr[kt] = load_a_frag(rowPtr, kt * 32, half, HID);

  // 32-bit byte offsets of the 8 (m = v + 8*half) gather rows this lane
  // covers, pre-biased by this lane's column (head*HID + r)
  int offL[8], offR[8];
  int colbias = (head * HID + r) * 4;
#pragma unroll
  for (int v = 0; v < 8; v++) {
    int e = eb + v + 8 * half;
    int s, d;
    if (e < N_EDGES) { s = src[e]; d = dst[e]; }
    else             { s = d = e - N_EDGES; }
    offL[v] = s * (HH * 4) + colbias;
    offR[v] = d * (HH * 4) + colbias;
  }
  const float* attp = att + head * HID + r;
  const __bf16* pb = WeP + (((size_t)head * 8) * 32 + lane) * 16;  // chunk/kt offsets immediate

  float part[8] = {0.f, 0.f, 0.f, 0.f, 0.f, 0.f, 0.f, 0.f};
  constexpr int Ntiles = HH >> 4;  // 32
#pragma unroll
  for (int chunk = 0; chunk < 8; chunk++) {
    // load all four B fragments up front (one clause), two acc chains
    bf16x16 b0 = *(const bf16x16*)(pb + ((size_t)0 * Ntiles + chunk) * 512);
    bf16x16 b1 = *(const bf16x16*)(pb + ((size_t)1 * Ntiles + chunk) * 512);
    bf16x16 b2 = *(const bf16x16*)(pb + ((size_t)2 * Ntiles + chunk) * 512);
    bf16x16 b3 = *(const bf16x16*)(pb + ((size_t)3 * Ntiles + chunk) * 512);
    // batched gathers: uniform base + 32-bit voffset + immediate (chunk*64B)
    float xlv[8], xrv[8];
#pragma unroll
    for (int v = 0; v < 8; v++)
      xlv[v] = *(const float*)((const char*)XL + offL[v] + chunk * 64);
#pragma unroll
    for (int v = 0; v < 8; v++)
      xrv[v] = *(const float*)((const char*)XR + offR[v] + chunk * 64);
    float attv = attp[chunk * 16];

    f32x8 acc0 = {}, acc1 = {};
    acc0 = __builtin_amdgcn_wmma_f32_16x16x32_bf16(false, afr[0], false, b0,
                                                   (short)0, acc0, false, false);
    acc1 = __builtin_amdgcn_wmma_f32_16x16x32_bf16(false, afr[1], false, b1,
                                                   (short)0, acc1, false, false);
    acc0 = __builtin_amdgcn_wmma_f32_16x16x32_bf16(false, afr[2], false, b2,
                                                   (short)0, acc0, false, false);
    acc1 = __builtin_amdgcn_wmma_f32_16x16x32_bf16(false, afr[3], false, b3,
                                                   (short)0, acc1, false, false);
#pragma unroll
    for (int v = 0; v < 8; v++) {
      float t = (acc0[v] + acc1[v]) + xlv[v] + xrv[v];
      t = (t > 0.f) ? t : NEGS * t;
      part[v] += t * attv;
    }
  }
  // reduce the 16 lanes of each half (rows 0-7 in lanes 0-15, rows 8-15 in 16-31)
#pragma unroll
  for (int v = 0; v < 8; v++) {
    float p = part[v];
    p += __shfl_xor(p, 1, 16);
    p += __shfl_xor(p, 2, 16);
    p += __shfl_xor(p, 4, 16);
    p += __shfl_xor(p, 8, 16);
    part[v] = p;
  }
  if (r == 0) {
#pragma unroll
    for (int v = 0; v < 8; v++) {
      int e = eb + v + 8 * half;
      logits[(size_t)e * HEADS + head] = part[v];
    }
  }
}

// ------------------------- softmax / aggregation ---------------------------
__global__ void k_fill(float* __restrict__ p, float v, int n) {
  int i = blockIdx.x * blockDim.x + threadIdx.x;
  if (i < n) p[i] = v;
}

__global__ void k_deg(const int* __restrict__ dst, float* __restrict__ deg) {
  int i = blockIdx.x * blockDim.x + threadIdx.x;
  if (i < N_EDGES) atomicAdd(&deg[dst[i]], 1.0f);
}

__global__ void k_eloop_acc(const float* __restrict__ ef, const int* __restrict__ dst,
                            float* __restrict__ eloop) {
  int e = blockIdx.x;
  int t = threadIdx.x;
  atomicAdd(&eloop[(size_t)dst[e] * HID + t], ef[(size_t)e * HID + t]);
}

__global__ void k_eloop_norm(float* __restrict__ eloop, const float* __restrict__ deg) {
  int i = blockIdx.x * blockDim.x + threadIdx.x;
  if (i < N_NODES * HID) eloop[i] /= fmaxf(deg[i / HID], 1.0f);
}

__global__ void k_segmax(const float* __restrict__ logits, const int* __restrict__ dst,
                         float* __restrict__ lmax) {
  int i = blockIdx.x * blockDim.x + threadIdx.x;
  if (i >= EF * HEADS) return;
  int e = i >> 2, h = i & 3;
  int d = (e < N_EDGES) ? dst[e] : (e - N_EDGES);
  atomicMax(&lmax[d * HEADS + h], logits[i]);
}

__global__ void k_pdenom(float* __restrict__ logits, const int* __restrict__ dst,
                         const float* __restrict__ lmax, float* __restrict__ denom) {
  int i = blockIdx.x * blockDim.x + threadIdx.x;
  if (i >= EF * HEADS) return;
  int e = i >> 2, h = i & 3;
  int d = (e < N_EDGES) ? dst[e] : (e - N_EDGES);
  float p = __expf(logits[i] - lmax[d * HEADS + h]);
  logits[i] = p;                         // logits buffer now holds p
  atomicAdd(&denom[d * HEADS + h], p);
}

__global__ void k_aggregate(const float* __restrict__ XL, const float* __restrict__ p,
                            const float* __restrict__ denom, const int* __restrict__ src,
                            const int* __restrict__ dst, float* __restrict__ outacc) {
  int e = blockIdx.x;
  int t = threadIdx.x;       // 128 threads, 4 cols each
  int s, d;
  if (e < N_EDGES) { s = src[e]; d = dst[e]; } else { s = d = e - N_EDGES; }
  int head = t >> 5;
  float alpha = p[(size_t)e * HEADS + head] / denom[(size_t)d * HEADS + head];
  f32x4 x = *(const f32x4*)(XL + (size_t)s * HH + t * 4);
  float* o = outacc + (size_t)d * HH + t * 4;
  atomicAdd(o + 0, alpha * x[0]);
  atomicAdd(o + 1, alpha * x[1]);
  atomicAdd(o + 2, alpha * x[2]);
  atomicAdd(o + 3, alpha * x[3]);
}

__global__ void k_finalize(const float* __restrict__ outacc, const float* __restrict__ bias,
                           float* __restrict__ hout) {
  int i = blockIdx.x * blockDim.x + threadIdx.x;
  if (i >= N_NODES * HID) return;
  int n = i / HID, c = i % HID;
  const float* o = outacc + (size_t)n * HH + c;
  float v = 0.25f * (o[0] + o[HID] + o[2 * HID] + o[3 * HID]) + bias[c];
  hout[i] = v > 0.f ? v : 0.f;
}

// ---------------------------------------------------------------------------
extern "C" void kernel_launch(void* const* d_in, const int* in_sizes, int n_in,
                              void* d_out, int out_size, void* d_ws, size_t ws_size,
                              hipStream_t stream) {
  const float* x      = (const float*)d_in[0];
  const int*   ei     = (const int*)  d_in[1];
  const float* eattr  = (const float*)d_in[2];
  const float* node_W = (const float*)d_in[3];
  const float* node_b = (const float*)d_in[4];
  const float* edge_W = (const float*)d_in[5];
  const float* edge_b = (const float*)d_in[6];
  const float* Wl     = (const float*)d_in[7];
  const float* bl     = (const float*)d_in[8];
  const float* Wr     = (const float*)d_in[9];
  const float* br     = (const float*)d_in[10];
  const float* We     = (const float*)d_in[11];
  const float* att    = (const float*)d_in[12];
  const float* bias   = (const float*)d_in[13];
  const int* src = ei;
  const int* dst = ei + N_EDGES;

  char* ws = (char*)d_ws;
  size_t off = 0;
  auto carve = [&](size_t bytes) -> char* {
    char* p = ws + off;
    off += (bytes + 255) & ~(size_t)255;
    return p;
  };
  float* hA     = (float*)carve((size_t)N_NODES * HID * 4);
  float* hB     = (float*)carve((size_t)N_NODES * HID * 4);
  float* ef     = (float*)carve((size_t)EF * HID * 4);
  float* XLb    = (float*)carve((size_t)N_NODES * HH * 4);
  float* XRb    = (float*)carve((size_t)N_NODES * HH * 4);
  float* outacc = (float*)carve((size_t)N_NODES * HH * 4);
  float* logits = (float*)carve((size_t)EF * HEADS * 4);
  float* lmax   = (float*)carve((size_t)N_NODES * HEADS * 4);
  float* denom  = (float*)carve((size_t)N_NODES * HEADS * 4);
  float* deg    = (float*)carve((size_t)N_NODES * 4);
  __bf16* nodeWp = (__bf16*)carve((size_t)2 * 8 * 512 * 2);
  __bf16* edgeWp = (__bf16*)carve((size_t)1 * 8 * 512 * 2);
  __bf16* WlP[3]; __bf16* WrP[3]; __bf16* WeP[3];
  for (int l = 0; l < 3; l++) {
    WlP[l] = (__bf16*)carve((size_t)4 * 32 * 512 * 2);
    WrP[l] = (__bf16*)carve((size_t)4 * 32 * 512 * 2);
    WeP[l] = (__bf16*)carve((size_t)4 * 32 * 512 * 2);
  }

  // ---- pack all weights to fragment-major bf16 (hot in 192MB L2) ----
  pack_b_bf16<<<dim3(2, 8),  32, 0, stream>>>(node_W, nodeWp, NODE_IN, HID);
  pack_b_bf16<<<dim3(1, 8),  32, 0, stream>>>(edge_W, edgeWp, EDGE_IN, HID);
  for (int l = 0; l < 3; l++) {
    pack_b_bf16<<<dim3(4, 32), 32, 0, stream>>>(Wl + (size_t)l * HID * HH, WlP[l], HID, HH);
    pack_b_bf16<<<dim3(4, 32), 32, 0, stream>>>(Wr + (size_t)l * HID * HH, WrP[l], HID, HH);
    pack_b_bf16<<<dim3(4, 32), 32, 0, stream>>>(We + (size_t)l * HID * HH, WeP[l], HID, HH);
  }

  // ---- embeddings: h0 = x@node_W + b ; e = edge_attr@edge_W + b ----
  gemm_bias_wmma<NODE_IN, HID, 8><<<dim3(N_NODES / 16, 1), 32, 0, stream>>>(
      x, nodeWp, node_b, hA, N_NODES);
  gemm_bias_wmma<EDGE_IN, HID, 8><<<dim3(N_EDGES / 16, 1), 32, 0, stream>>>(
      eattr, edgeWp, edge_b, ef, N_EDGES);

  // ---- self-loop edge attrs: e_loop[n] = mean of incoming e over dst ----
  k_fill<<<(N_NODES + 255) / 256, 256, 0, stream>>>(deg, 0.f, N_NODES);
  k_fill<<<(N_NODES * HID + 255) / 256, 256, 0, stream>>>(ef + (size_t)N_EDGES * HID, 0.f, N_NODES * HID);
  k_deg<<<(N_EDGES + 255) / 256, 256, 0, stream>>>(dst, deg);
  k_eloop_acc<<<N_EDGES, HID, 0, stream>>>(ef, dst, ef + (size_t)N_EDGES * HID);
  k_eloop_norm<<<(N_NODES * HID + 255) / 256, 256, 0, stream>>>(ef + (size_t)N_EDGES * HID, deg);

  // ---- 3 GATv2 layers ----
  float* hcur = hA;
  for (int l = 0; l < 3; l++) {
    float* hout = (l == 2) ? (float*)d_out : ((l == 0) ? hB : hA);

    gemm_bias_wmma<HID, HH, 8><<<dim3(N_NODES / 16, HH / 128), 32, 0, stream>>>(
        hcur, WlP[l], bl + (size_t)l * HH, XLb, N_NODES);
    gemm_bias_wmma<HID, HH, 8><<<dim3(N_NODES / 16, HH / 128), 32, 0, stream>>>(
        hcur, WrP[l], br + (size_t)l * HH, XRb, N_NODES);

    k_fill<<<(N_NODES * HEADS + 255) / 256, 256, 0, stream>>>(lmax, -3.0e38f, N_NODES * HEADS);
    k_fill<<<(N_NODES * HEADS + 255) / 256, 256, 0, stream>>>(denom, 0.f, N_NODES * HEADS);
    k_fill<<<(N_NODES * HH + 255) / 256, 256, 0, stream>>>(outacc, 0.f, N_NODES * HH);

    edge_logits_wmma<<<dim3(EF / 16, HEADS), 32, 0, stream>>>(
        ef, WeP[l], XLb, XRb, att + (size_t)l * HEADS * HID, src, dst, logits);

    k_segmax<<<(EF * HEADS + 255) / 256, 256, 0, stream>>>(logits, dst, lmax);
    k_pdenom<<<(EF * HEADS + 255) / 256, 256, 0, stream>>>(logits, dst, lmax, denom);
    k_aggregate<<<EF, HID, 0, stream>>>(XLb, logits, denom, src, dst, outacc);
    k_finalize<<<(N_NODES * HID + 255) / 256, 256, 0, stream>>>(outacc, bias + (size_t)l * HID, hout);

    hcur = hout;
  }
  (void)in_sizes; (void)n_in; (void)out_size; (void)ws_size;
}
